// SelectiveTokenMixer_72232759984889
// MI455X (gfx1250) — matrix operand order repaired and checked by reading, not compiled, not run on hardware
//
#include <hip/hip_runtime.h>
#include <hip/hip_bf16.h>
#include <stdint.h>

// ---------------------------------------------------------------------------
// Problem constants (from reference)
// ---------------------------------------------------------------------------
#define D_MODEL 768
#define D_STATE 16
#define D_CONV  4
#define D_INNER 1536           // EXPAND * D_MODEL
#define DT_RANK 48
#define BATCH   8
#define SEQ     1024
#define NTOK    (BATCH * SEQ)  // 8192 rows for all GEMMs

typedef __attribute__((ext_vector_type(16))) __bf16 v16bf;
typedef __attribute__((ext_vector_type(8)))  float  v8f;
typedef __attribute__((ext_vector_type(2)))  int    v2i;

union Frag16 {           // one WMMA 16-bit A/B operand: 16 bf16 per lane
    v16bf    v;
    uint4    q[2];       // two contiguous 16B runs -> 2x ds_load_b128
};

__device__ __forceinline__ float sigmoid_f(float v) { return 1.f / (1.f + __expf(-v)); }

// ---------------------------------------------------------------------------
// CDNA5 async global->LDS path (ASYNCcnt) with graceful fallback
// Builtin signature (from clang diagnostic): (V2i addrspace(1)*, V2i addrspace(3)*, Ii, Ii)
// ---------------------------------------------------------------------------
#if defined(__has_builtin)
#if __has_builtin(__builtin_amdgcn_global_load_async_to_lds_b64)
#define HAVE_ASYNC_LDS 1
#endif
#endif

typedef __attribute__((address_space(1))) v2i as1_v2i;
typedef __attribute__((address_space(3))) v2i as3_v2i;

__device__ __forceinline__ void async_copy_b64(const __bf16* gsrc, __bf16* ldst) {
#if defined(HAVE_ASYNC_LDS)
    __builtin_amdgcn_global_load_async_to_lds_b64(
        (as1_v2i*)(size_t)gsrc, (as3_v2i*)(v2i*)ldst, 0, 0);
#else
    *(uint2*)ldst = *(const uint2*)gsrc;
#endif
}

__device__ __forceinline__ void wait_async_all() {
#if defined(HAVE_ASYNC_LDS)
#if __has_builtin(__builtin_amdgcn_s_wait_asynccnt)
    __builtin_amdgcn_s_wait_asynccnt(0);
#else
    asm volatile("s_wait_asynccnt 0x0" ::: "memory");
#endif
#endif
}

// ---------------------------------------------------------------------------
// fp32 -> bf16 cast (weights / activations)
// ---------------------------------------------------------------------------
__global__ void cvt_f32_bf16(const float* __restrict__ in, __bf16* __restrict__ out, int n) {
    int i = blockIdx.x * blockDim.x + threadIdx.x;
    if (i < n) out[i] = (__bf16)in[i];
}

// ---------------------------------------------------------------------------
// LayerNorm over D_MODEL=768 (one block of 256 per row), bf16 output for WMMA
// ---------------------------------------------------------------------------
__global__ __launch_bounds__(256) void ln_kernel(const float* __restrict__ x,
                                                 const float* __restrict__ g,
                                                 const float* __restrict__ b,
                                                 __bf16* __restrict__ xn) {
    const int row = blockIdx.x;
    const int tid = threadIdx.x;
    const float* xr = x + (size_t)row * D_MODEL;
    float v0 = xr[tid], v1 = xr[tid + 256], v2 = xr[tid + 512];
    float s  = v0 + v1 + v2;
    float sq = v0 * v0 + v1 * v1 + v2 * v2;
    #pragma unroll
    for (int off = 16; off > 0; off >>= 1) {
        s  += __shfl_down(s,  off, 32);
        sq += __shfl_down(sq, off, 32);
    }
    __shared__ float ws[8], wq[8];
    __shared__ float mu_s, rs_s;
    if ((tid & 31) == 0) { ws[tid >> 5] = s; wq[tid >> 5] = sq; }
    __syncthreads();
    if (tid == 0) {
        float S = 0.f, Q = 0.f;
        #pragma unroll
        for (int i = 0; i < 8; ++i) { S += ws[i]; Q += wq[i]; }
        float mu  = S / (float)D_MODEL;
        float var = Q / (float)D_MODEL - mu * mu;
        mu_s = mu;
        rs_s = rsqrtf(var + 1e-5f);
    }
    __syncthreads();
    float mu = mu_s, rs = rs_s;
    __bf16* xo = xn + (size_t)row * D_MODEL;
    xo[tid]       = (__bf16)((v0 - mu) * rs * g[tid]       + b[tid]);
    xo[tid + 256] = (__bf16)((v1 - mu) * rs * g[tid + 256] + b[tid + 256]);
    xo[tid + 512] = (__bf16)((v2 - mu) * rs * g[tid + 512] + b[tid + 512]);
}

// ---------------------------------------------------------------------------
// General bf16 WMMA GEMM: C(MxN) = A(MxK) * B(KxN), f32 accumulate.
// Block = 256 threads = 8 waves; block tile 128x128, K-step 32.
// Double-buffered LDS; A tile streamed via GLOBAL_LOAD_ASYNC_TO_LDS_B64
// (ASYNCcnt + s_wait_asynccnt); B tile pipelined through registers and
// transpose-stored to LDS after the current tile's fragment ds_loads.
// Fragment reads are 2x ds_load_b128 per operand (contiguous 16B runs).
// Wave w -> (wr=w>>1, wc=w&1): 32x64 region = 2x4 subtiles of 16x16.
// ---------------------------------------------------------------------------
#define BM  128
#define BN  128
#define BK  32
#define LDA 40   // bf16 units; 80B rows: 16B-aligned b128 frag reads, conflict-free
#define LDB 40

enum { EPI_NONE = 0, EPI_AUX = 1, EPI_SOFTPLUS = 2, EPI_RESID = 3 };

template <int EPI>
__global__ __launch_bounds__(256) void wmma_gemm_bf16(
    const __bf16* __restrict__ A, const __bf16* __restrict__ Bmat,
    float* __restrict__ C, int M, int N, int K,
    const float* __restrict__ bias, const float* __restrict__ resid,
    __bf16* __restrict__ aux, int auxN) {
    __shared__ __bf16 lds_a[2][BM * LDA];
    __shared__ __bf16 lds_b[2][BN * LDB];   // B tile stored transposed: [n][k]

    const int tid  = threadIdx.x;
    const int lane = tid & 31;
    const int wave = tid >> 5;
    const int wr   = wave >> 1;   // 0..3
    const int wc   = wave & 1;    // 0..1
    const int hi   = lane >> 4;   // half-wave select
    const int ml   = lane & 15;
    const int blockN = blockIdx.x * BN;
    const int blockM = blockIdx.y * BM;

    v8f acc[2][4];
    #pragma unroll
    for (int i = 0; i < 2; ++i)
        #pragma unroll
        for (int j = 0; j < 4; ++j)
            #pragma unroll
            for (int r = 0; r < 8; ++r) acc[i][j][r] = 0.f;

    uint2 breg[4];   // B-tile staging registers (4 x 8B per thread)

    // ---- issue async A-tile loads (128 rows x 32 cols, 8B chunks) ----
    auto issue_a = [&](int k0, int buf) {
        #pragma unroll
        for (int it = 0; it < 4; ++it) {
            int task = tid + it * 256;          // BM*(BK/4) = 1024 tasks
            int row = task >> 3, cg = task & 7;
            int gm = blockM + row;
            int gk = k0 + cg * 4;
            __bf16* dst = &lds_a[buf][row * LDA + cg * 4];
            if (gm < M && gk + 4 <= K) {
                async_copy_b64(A + (size_t)gm * K + gk, dst);
            } else {
                #pragma unroll
                for (int e = 0; e < 4; ++e) {
                    __bf16 v = (__bf16)0.f;
                    if (gm < M && gk + e < K) v = A[(size_t)gm * K + gk + e];
                    dst[e] = v;
                }
            }
        }
    };
    // ---- B tile: global -> regs (issue early) ----
    auto load_b = [&](int k0) {
        #pragma unroll
        for (int it = 0; it < 4; ++it) {
            int task = tid + it * 256;          // BK*(BN/4) = 1024 tasks
            int krow = task >> 5, ng = task & 31;
            int gk = k0 + krow;
            int gn = blockN + ng * 4;
            if (gk < K && gn + 4 <= N) {
                breg[it] = *(const uint2*)(Bmat + (size_t)gk * N + gn);
            } else {
                __bf16 tmp[4];
                #pragma unroll
                for (int e = 0; e < 4; ++e)
                    tmp[e] = (gk < K && gn + e < N) ? Bmat[(size_t)gk * N + gn + e]
                                                    : (__bf16)0.f;
                breg[it] = *(uint2*)tmp;
            }
        }
    };
    // ---- B tile: regs -> LDS transposed (after current tile's ds_loads) ----
    auto store_b = [&](int buf) {
        #pragma unroll
        for (int it = 0; it < 4; ++it) {
            int task = tid + it * 256;
            int krow = task >> 5, ng = task & 31;
            __bf16 tmp[4];
            *(uint2*)tmp = breg[it];
            #pragma unroll
            for (int e = 0; e < 4; ++e)
                lds_b[buf][(ng * 4 + e) * LDB + krow] = tmp[e];
        }
    };

    const int ntiles = (K + BK - 1) / BK;

    // prologue: tile 0 into buffer 0
    issue_a(0, 0);
    load_b(0);
    store_b(0);

    for (int kt = 0; kt < ntiles; ++kt) {
        const int cur = kt & 1;
        wait_async_all();       // our async writes of tile kt have landed in LDS
        __syncthreads();        // everyone's writes visible; tile kt ready

        if (kt + 1 < ntiles) {
            const int k0n = (kt + 1) * BK;
            issue_a(k0n, cur ^ 1);      // async: overlaps with WMMA below
            load_b(k0n);                // global->regs: overlaps with WMMA below
            if (kt + 2 < ntiles) {      // speculative prefetch two tiles ahead
                __builtin_prefetch(&A[(size_t)(blockM + (tid & (BM - 1))) * K +
                                      (kt + 2) * BK], 0, 1);
                __builtin_prefetch(&Bmat[(size_t)((kt + 2) * BK + (tid & (BK - 1))) * N +
                                         blockN], 0, 1);
            }
        }

        // ---- fragments per the CDNA5 wave32 16-bit layouts ----
        // A lane (m, hi): K runs [8hi .. 8hi+7] and [16+8hi .. 16+8hi+7]  (2x16B)
        // B lane (n, hi): K runs [8hi .. 8hi+7] and [16+8hi .. 16+8hi+7]  (2x16B)
        Frag16 af[2], bf[4];
        #pragma unroll
        for (int i = 0; i < 2; ++i) {
            const __bf16* ap = &lds_a[cur][(wr * 32 + i * 16 + ml) * LDA];
            af[i].q[0] = *(const uint4*)(ap + hi * 8);
            af[i].q[1] = *(const uint4*)(ap + 16 + hi * 8);
        }
        #pragma unroll
        for (int j = 0; j < 4; ++j) {
            const __bf16* bp = &lds_b[cur][(wc * 64 + j * 16 + ml) * LDB];
            bf[j].q[0] = *(const uint4*)(bp + hi * 8);
            bf[j].q[1] = *(const uint4*)(bp + 16 + hi * 8);
        }
        #pragma unroll
        for (int i = 0; i < 2; ++i)
            #pragma unroll
            for (int j = 0; j < 4; ++j)
                acc[i][j] = __builtin_amdgcn_wmma_f32_16x16x32_bf16(
                    false, af[i].v, false, bf[j].v, (short)0, acc[i][j], false, false);

        if (kt + 1 < ntiles) store_b(cur ^ 1);  // ds stores after frag ds_loads
    }

    // ---- store (C/D layout: VGPR r -> M = r + 8*hi, N = lane&15) ----
    #pragma unroll
    for (int i = 0; i < 2; ++i) {
        #pragma unroll
        for (int j = 0; j < 4; ++j) {
            #pragma unroll
            for (int r = 0; r < 8; ++r) {
                int gm = blockM + wr * 32 + i * 16 + hi * 8 + r;
                int gn = blockN + wc * 64 + j * 16 + ml;
                if (gm < M && gn < N) {
                    float v = acc[i][j][r];
                    if (EPI == EPI_SOFTPLUS) {
                        v += bias[gn];
                        v = (v > 20.f) ? v : log1pf(__expf(v));
                        C[(size_t)gm * N + gn] = v;
                    } else if (EPI == EPI_RESID) {
                        C[(size_t)gm * N + gn] = v + resid[(size_t)gm * N + gn];
                    } else {
                        C[(size_t)gm * N + gn] = v;
                        if (EPI == EPI_AUX && gn < auxN)
                            aux[(size_t)gm * auxN + gn] = (__bf16)v;
                    }
                }
            }
        }
    }
}

// ---------------------------------------------------------------------------
// Causal depthwise conv (taps=4) + SiLU; reads xp = xz[:, 0:1536]
// writes fp32 (for scan / Dskip) and bf16 (for the W_x GEMM)
// ---------------------------------------------------------------------------
__global__ __launch_bounds__(256) void conv_silu_kernel(
    const float* __restrict__ xz, const float* __restrict__ conv_w,
    const float* __restrict__ conv_b, float* __restrict__ xc_f,
    __bf16* __restrict__ xc_bf) {
    int idx = blockIdx.x * blockDim.x + threadIdx.x;
    if (idx >= NTOK * D_INNER) return;
    int d = idx % D_INNER;
    int r = idx / D_INNER;          // r = b*SEQ + t
    int t = r % SEQ;
    float acc = conv_b[d];
    #pragma unroll
    for (int j = 0; j < D_CONV; ++j) {
        int ts = t - (D_CONV - 1) + j;
        if (ts >= 0)
            acc += conv_w[d * D_CONV + j] *
                   xz[(size_t)(r - (D_CONV - 1) + j) * (2 * D_INNER) + d];
    }
    float v = acc * sigmoid_f(acc);
    xc_f[idx]  = v;
    xc_bf[idx] = (__bf16)v;
}

// ---------------------------------------------------------------------------
// Selective (S6) scan fused with Dskip and SiLU(z) gate.
// One thread per (b, d) channel; h[16] in registers; B/C staged in LDS
// in chunks of 64 timesteps (shared by all 256 channels in the block).
// Output y_gated as bf16 (A operand of the W_out GEMM).
// ---------------------------------------------------------------------------
#define SCAN_CHUNK 64

__global__ __launch_bounds__(256) void scan_kernel(
    const float* __restrict__ xdbl,   // (NTOK, 80): [48:64]=B, [64:80]=C
    const float* __restrict__ dt,     // (NTOK, D_INNER)
    const float* __restrict__ xc,     // (NTOK, D_INNER)
    const float* __restrict__ xz,     // (NTOK, 2*D_INNER): gate z = cols 1536+
    const float* __restrict__ A_log,  // (D_INNER, 16)
    const float* __restrict__ Dskip,  // (D_INNER)
    __bf16* __restrict__ ybf)         // (NTOK, D_INNER)
{
    const int d = blockIdx.x * blockDim.x + threadIdx.x;  // 0..1535
    const int b = blockIdx.y;
    __shared__ float sBC[SCAN_CHUNK][32];

    float A[D_STATE];
    #pragma unroll
    for (int n = 0; n < D_STATE; ++n) A[n] = -__expf(A_log[d * D_STATE + n]);
    const float Dv = Dskip[d];

    float h[D_STATE];
    #pragma unroll
    for (int n = 0; n < D_STATE; ++n) h[n] = 0.f;

    for (int t0 = 0; t0 < SEQ; t0 += SCAN_CHUNK) {
        __syncthreads();
        for (int i = threadIdx.x; i < SCAN_CHUNK * 32; i += blockDim.x) {
            int tt = i >> 5, j = i & 31;
            sBC[tt][j] = xdbl[(size_t)(b * SEQ + t0 + tt) * 80 + 48 + j];
        }
        __syncthreads();
        for (int tt = 0; tt < SCAN_CHUNK; ++tt) {
            size_t r = (size_t)(b * SEQ + t0 + tt);
            float xv  = xc[r * D_INNER + d];
            float dtv = dt[r * D_INNER + d];
            float dx  = dtv * xv;
            float y = 0.f;
            #pragma unroll
            for (int n = 0; n < D_STATE; ++n) {
                float dA = __expf(dtv * A[n]);
                h[n] = dA * h[n] + dx * sBC[tt][n];
                y += h[n] * sBC[tt][16 + n];
            }
            float zv = xz[r * (2 * D_INNER) + D_INNER + d];
            float yg = (y + Dv * xv) * (zv * sigmoid_f(zv));
            ybf[r * D_INNER + d] = (__bf16)yg;
        }
    }
}

// ---------------------------------------------------------------------------
// Host-side launcher
// ---------------------------------------------------------------------------
extern "C" void kernel_launch(void* const* d_in, const int* in_sizes, int n_in,
                              void* d_out, int out_size, void* d_ws, size_t ws_size,
                              hipStream_t stream) {
    const float* x      = (const float*)d_in[0];
    const float* ln_g   = (const float*)d_in[1];
    const float* ln_b   = (const float*)d_in[2];
    const float* W_in   = (const float*)d_in[3];
    const float* conv_w = (const float*)d_in[4];
    const float* conv_b = (const float*)d_in[5];
    const float* W_x    = (const float*)d_in[6];
    const float* W_dt   = (const float*)d_in[7];
    const float* b_dt   = (const float*)d_in[8];
    const float* A_log  = (const float*)d_in[9];
    const float* Dskip  = (const float*)d_in[10];
    const float* W_out  = (const float*)d_in[11];
    float* out = (float*)d_out;

    char* ws = (char*)d_ws;
    size_t off = 0;
    auto carve = [&](size_t bytes) -> void* {
        void* p = ws + off;
        off = (off + bytes + 255) & ~(size_t)255;
        return p;
    };

    __bf16* Win_bf  = (__bf16*)carve((size_t)D_MODEL * 2 * D_INNER * 2);
    __bf16* Wx_bf   = (__bf16*)carve((size_t)D_INNER * (DT_RANK + 2 * D_STATE) * 2);
    __bf16* Wdt_bf  = (__bf16*)carve((size_t)DT_RANK * D_INNER * 2);
    __bf16* Wout_bf = (__bf16*)carve((size_t)D_INNER * D_MODEL * 2);
    __bf16* xn_bf   = (__bf16*)carve((size_t)NTOK * D_MODEL * 2);
    float*  xz      = (float*)carve((size_t)NTOK * 2 * D_INNER * 4);
    float*  xc_f    = (float*)carve((size_t)NTOK * D_INNER * 4);
    __bf16* xc_bf   = (__bf16*)carve((size_t)NTOK * D_INNER * 2);
    float*  xdbl    = (float*)carve((size_t)NTOK * (DT_RANK + 2 * D_STATE) * 4);
    __bf16* dtA_bf  = (__bf16*)carve((size_t)NTOK * DT_RANK * 2);
    float*  dt_f    = (float*)carve((size_t)NTOK * D_INNER * 4);
    __bf16* y_bf    = (__bf16*)carve((size_t)NTOK * D_INNER * 2);

    // 1) weight casts to bf16
    {
        int n1 = D_MODEL * 2 * D_INNER;
        cvt_f32_bf16<<<(n1 + 255) / 256, 256, 0, stream>>>(W_in, Win_bf, n1);
        int n2 = D_INNER * (DT_RANK + 2 * D_STATE);
        cvt_f32_bf16<<<(n2 + 255) / 256, 256, 0, stream>>>(W_x, Wx_bf, n2);
        int n3 = DT_RANK * D_INNER;
        cvt_f32_bf16<<<(n3 + 255) / 256, 256, 0, stream>>>(W_dt, Wdt_bf, n3);
        int n4 = D_INNER * D_MODEL;
        cvt_f32_bf16<<<(n4 + 255) / 256, 256, 0, stream>>>(W_out, Wout_bf, n4);
    }

    // 2) LayerNorm -> xn (bf16)
    ln_kernel<<<NTOK, 256, 0, stream>>>(x, ln_g, ln_b, xn_bf);

    // 3) xz = xn @ W_in   (8192 x 3072, K=768)
    {
        dim3 grid((2 * D_INNER + BN - 1) / BN, (NTOK + BM - 1) / BM);
        wmma_gemm_bf16<EPI_NONE><<<grid, 256, 0, stream>>>(
            xn_bf, Win_bf, xz, NTOK, 2 * D_INNER, D_MODEL, nullptr, nullptr, nullptr, 0);
    }

    // 4) causal depthwise conv + SiLU -> xc (fp32 + bf16)
    {
        int n = NTOK * D_INNER;
        conv_silu_kernel<<<(n + 255) / 256, 256, 0, stream>>>(xz, conv_w, conv_b, xc_f, xc_bf);
    }

    // 5) x_dbl = xc @ W_x (8192 x 80, K=1536); aux: bf16 copy of dt-rank cols
    {
        dim3 grid((DT_RANK + 2 * D_STATE + BN - 1) / BN, (NTOK + BM - 1) / BM);
        wmma_gemm_bf16<EPI_AUX><<<grid, 256, 0, stream>>>(
            xc_bf, Wx_bf, xdbl, NTOK, DT_RANK + 2 * D_STATE, D_INNER,
            nullptr, nullptr, dtA_bf, DT_RANK);
    }

    // 6) dt = softplus(dtA @ W_dt + b_dt)  (8192 x 1536, K=48)
    {
        dim3 grid((D_INNER + BN - 1) / BN, (NTOK + BM - 1) / BM);
        wmma_gemm_bf16<EPI_SOFTPLUS><<<grid, 256, 0, stream>>>(
            dtA_bf, Wdt_bf, dt_f, NTOK, D_INNER, DT_RANK, b_dt, nullptr, nullptr, 0);
    }

    // 7) selective scan + Dskip + SiLU(z) gate -> y (bf16)
    {
        dim3 grid(D_INNER / 256, BATCH);
        scan_kernel<<<grid, 256, 0, stream>>>(xdbl, dt_f, xc_f, xz, A_log, Dskip, y_bf);
    }

    // 8) out = y @ W_out + x  (8192 x 768, K=1536)
    {
        dim3 grid((D_MODEL + BN - 1) / BN, (NTOK + BM - 1) / BM);
        wmma_gemm_bf16<EPI_RESID><<<grid, 256, 0, stream>>>(
            y_bf, Wout_bf, out, NTOK, D_MODEL, D_INNER, nullptr, x, nullptr, 0);
    }
}
